// BG_LPS_82695300317254
// MI455X (gfx1250) — matrix operand order repaired and checked by reading, not compiled, not run on hardware
//
#include <hip/hip_runtime.h>
#include <stdint.h>

// ---------------- model dims (match reference) ----------------
#define NPA   65536
#define NLA   8192
#define NBC   512          // batch of complexes
#define E_PP  131072
#define E_LL  16384
#define E_LP  65536
#define E_PL  65536
#define CH    128          // hidden channels
#define HDS   4            // attention heads
#define HC    512          // HDS*CH
#define G3    384          // 3*CH (GRU gates)
#define KIN   39           // NB_ATOM
#define KINP  64           // KIN padded to 32-multiple
#define KE    134          // CH + EDGE_DIM
#define KEP   160          // KE padded
#define NEGS  0.2f

typedef unsigned short u16;

// ---------------- WMMA types ----------------
typedef __attribute__((ext_vector_type(16))) __bf16 bf16x16;
typedef __attribute__((ext_vector_type(8)))  float  f32x8;

union BfFrag { bf16x16 v; u16 u[16]; uint4 q[2]; };

__device__ __forceinline__ u16 f2bf(float f){
  unsigned x = __float_as_uint(f);
  return (u16)((x + 0x7fffu + ((x >> 16) & 1u)) >> 16);  // RNE to bf16
}
__device__ __forceinline__ f32x8 wmma_bf16(BfFrag a, BfFrag b, f32x8 c){
  return __builtin_amdgcn_wmma_f32_16x16x32_bf16(false, a.v, false, b.v, (short)0, c, false, false);
}
__device__ __forceinline__ float lreluf(float v){ return v > 0.f ? v : NEGS * v; }
__device__ __forceinline__ float eluf(float v){ return v > 0.f ? v : expm1f(v); }
__device__ __forceinline__ float sigmf(float v){ return 1.f / (1.f + expf(-v)); }
// monotone float<->uint map for atomic segment-max
__device__ __forceinline__ unsigned ou_enc(float f){
  unsigned b = __float_as_uint(f);
  return (b & 0x80000000u) ? ~b : (b | 0x80000000u);
}
__device__ __forceinline__ float ou_dec(unsigned u){
  unsigned b = (u & 0x80000000u) ? (u & 0x7fffffffu) : ~u;
  return __uint_as_float(b);
}

// ============================================================
// Packing kernels (run once per launch; weights are tiny).
// WT layout: [M][Kp] bf16 (transposed, K zero-padded to Kp).
// ============================================================
__global__ void k_pack_wT(const float* __restrict__ W, u16* __restrict__ WT,
                          int K, int M, int Kp){
  long i = (long)blockIdx.x * blockDim.x + threadIdx.x;
  if (i < (long)M * Kp){
    const int m = (int)(i / Kp), kp = (int)(i % Kp);
    WT[i] = (kp < K) ? f2bf(W[(long)kp * M + m]) : (u16)0;
  }
}
// pack rows of X[N,K] into Xb[N,Kp] bf16 (zero pad)
__global__ void k_pack_rows(const float* __restrict__ X, u16* __restrict__ Xb,
                            int K, int Kp, long N){
  long i = (long)blockIdx.x * blockDim.x + threadIdx.x;
  if (i < N * Kp){
    const long n = i / Kp; const int kp = (int)(i % Kp);
    Xb[i] = (kp < K) ? f2bf(X[n * K + kp]) : (u16)0;
  }
}

// ============================================================
// Dense GEMM: Y[N,M] = act(Xb[N,Kp](bf16) @ WT[M,Kp]^T(bf16) + bias)
// One wave per 16x64 output strip (4 WMMA tiles share A fragment).
// No guards: N%16==0, M%64==0, Kp%32==0 by construction.
// act: 0 none, 1 leaky_relu(0.2), 2 relu.  Yb: optional bf16 mirror.
// ============================================================
__global__ __launch_bounds__(32) void k_gemm_bf(
    const u16* __restrict__ Xb, const u16* __restrict__ WT,
    const float* __restrict__ bias, float* __restrict__ Y,
    u16* __restrict__ Yb, int Kp, int M, int act)
{
  const int lane = threadIdx.x;
  const int l15 = lane & 15, hi = lane >> 4;
  const int row = blockIdx.x * 16 + l15;
  const int colbase = blockIdx.y * 64;
  const u16* xr = Xb + (size_t)row * Kp;
  const u16* wc[4];
#pragma unroll
  for (int t = 0; t < 4; ++t) wc[t] = WT + (size_t)(colbase + t * 16 + l15) * Kp;
  f32x8 acc[4] = {{}, {}, {}, {}};
  for (int k0 = 0; k0 < Kp; k0 += 32){
    BfFrag a;
    const int ka = k0 + hi * 16;
    a.q[0] = *(const uint4*)(xr + ka);
    a.q[1] = *(const uint4*)(xr + ka + 8);
    const int kb = k0 + hi * 16;
#pragma unroll
    for (int t = 0; t < 4; ++t){
      BfFrag b;
      b.q[0] = *(const uint4*)(wc[t] + kb);
      b.q[1] = *(const uint4*)(wc[t] + kb + 8);
      acc[t] = wmma_bf16(a, b, acc[t]);
    }
  }
#pragma unroll
  for (int t = 0; t < 4; ++t){
    const int col = colbase + t * 16 + l15;
    const float bv = bias ? bias[col] : 0.f;
#pragma unroll
    for (int r = 0; r < 8; ++r){
      const int orow = blockIdx.x * 16 + r + hi * 8;
      float v = acc[t][r] + bv;
      if (act == 1) v = lreluf(v);
      else if (act == 2) v = fmaxf(v, 0.f);
      Y[(size_t)orow * M + col] = v;
      if (Yb) Yb[(size_t)orow * M + col] = f2bf(v);
    }
  }
}

// ============================================================
// GATEConv edge kernel, tile of 16 edges, 8 waves (one per 16 cols):
//   m  = lrelu([x1s[src] ‖ ea] @ W1)   (bf16 WMMA, K=160 padded)
//   a  = lrelu(m . att_l + AR[dst])    -> Alog[e]
//   m2 = m @ W2                        -> M2[e,128]
// x1s comes in as bf16 mirror; W1T/W2T pre-transposed bf16.
// ============================================================
__global__ __launch_bounds__(256) void k_gate_edge(
    const u16* __restrict__ X1b, const int* __restrict__ src,
    const int* __restrict__ dst, const float* __restrict__ EA,
    const u16* __restrict__ W1T, const u16* __restrict__ W2T,
    const float* __restrict__ att_l, const float* __restrict__ AR,
    float* __restrict__ Alog, float* __restrict__ M2)
{
  __shared__ float mS[16][CH + 4];                 // f32 m (attention dot)
  __shared__ __align__(16) u16 mB[16][CH];         // bf16 m (2nd WMMA pass)
  __shared__ int   sS[16], dS[16];
  __shared__ float eS[16][6];
  const int tid  = threadIdx.x;
  const int wv   = tid >> 5;
  const int lane = tid & 31;
  const int eb   = blockIdx.x * 16;
  if (tid < 16){ sS[tid] = src[eb + tid]; dS[tid] = dst[eb + tid]; }
  if (tid < 96){ const int i = tid / 6, j = tid % 6; eS[i][j] = EA[(long)(eb + i) * 6 + j]; }
  __syncthreads();

  const int r16 = lane & 15;
  const int hi  = lane >> 4;
  const int col = wv * 16 + r16;

  // ---- m = lrelu(concat @ W1) ----
  f32x8 acc = {};
  const u16* xr = X1b + (size_t)sS[r16] * CH;
  const u16* w1 = W1T + (size_t)col * KEP;
#pragma unroll
  for (int kc = 0; kc < 4; ++kc){                   // K 0..127 from x1s[src]
    const int k0 = kc << 5;
    BfFrag a, b;
    const int ka = k0 + hi * 16;
    a.q[0] = *(const uint4*)(xr + ka);
    a.q[1] = *(const uint4*)(xr + ka + 8);
    const int kb = k0 + hi * 16;
    b.q[0] = *(const uint4*)(w1 + kb);
    b.q[1] = *(const uint4*)(w1 + kb + 8);
    acc = wmma_bf16(a, b, acc);
  }
  {                                                 // K 128..159: only 128..133 nonzero (edge feats)
    BfFrag a, b;
#pragma unroll
    for (int j = 0; j < 16; ++j) a.u[j] = 0;
    if (hi == 0){
#pragma unroll
      for (int j = 0; j < 6; ++j) a.u[j] = f2bf(eS[r16][j]);
    }
    const int kb = 128 + hi * 16;
    b.q[0] = *(const uint4*)(w1 + kb);
    b.q[1] = *(const uint4*)(w1 + kb + 8);
    acc = wmma_bf16(a, b, acc);
  }
#pragma unroll
  for (int r = 0; r < 8; ++r){
    const float v = lreluf(acc[r]);
    mS[r + hi * 8][col] = v;
    mB[r + hi * 8][col] = f2bf(v);
  }
  __syncthreads();

  // ---- attention logit per edge ----
  if (tid < 16){
    float s = 0.f;
#pragma unroll 4
    for (int c = 0; c < CH; ++c) s += mS[tid][c] * att_l[c];
    Alog[eb + tid] = lreluf(s + AR[dS[tid]]);
  }

  // ---- m2 = m @ W2 (K=128, A from LDS bf16) ----
  f32x8 acc2 = {};
  const u16* w2 = W2T + (size_t)col * CH;
#pragma unroll
  for (int kc = 0; kc < 4; ++kc){
    const int k0 = kc << 5;
    BfFrag a, b;
    const int ka = k0 + hi * 16;
    a.q[0] = *(const uint4*)&mB[r16][ka];
    a.q[1] = *(const uint4*)&mB[r16][ka + 8];
    const int kb = k0 + hi * 16;
    b.q[0] = *(const uint4*)(w2 + kb);
    b.q[1] = *(const uint4*)(w2 + kb + 8);
    acc2 = wmma_bf16(a, b, acc2);
  }
#pragma unroll
  for (int r = 0; r < 8; ++r){
    const int e = eb + r + hi * 8;
    M2[(size_t)e * CH + col] = acc2[r];
  }
}

// ---------------- elementwise / segment kernels ----------------
__global__ void k_fill_f32(float* p, float v, long n){
  long i = (long)blockIdx.x * blockDim.x + threadIdx.x; if (i < n) p[i] = v;
}
__global__ void k_fill_u32(unsigned* p, unsigned v, long n){
  long i = (long)blockIdx.x * blockDim.x + threadIdx.x; if (i < n) p[i] = v;
}
__global__ void k_node_dot(const float* __restrict__ X, const float* __restrict__ w,
                           float* __restrict__ o, int N){
  int n = blockIdx.x * blockDim.x + threadIdx.x;
  if (n < N){
    const float* x = X + (long)n * CH;
    float s = 0.f;
#pragma unroll 4
    for (int c = 0; c < CH; ++c) s += x[c] * w[c];
    o[n] = s;
  }
}
__global__ void k_att_dot(const float* __restrict__ Q, const float* __restrict__ att,
                          float* __restrict__ o, int N){
  long i = (long)blockIdx.x * blockDim.x + threadIdx.x;
  if (i < (long)N * HDS){
    const int n = (int)(i >> 2), h = (int)(i & 3);
    const float* q = Q + (long)n * HC + h * CH;
    const float* a = att + h * CH;
    float s = 0.f;
#pragma unroll 4
    for (int c = 0; c < CH; ++c) s += q[c] * a[c];
    o[i] = s;
  }
}
__global__ void k_gat_logit(const float* __restrict__ SSp, const float* __restrict__ SDp,
                            const int* __restrict__ src, const int* __restrict__ dst,
                            float* __restrict__ A, int E){
  long i = (long)blockIdx.x * blockDim.x + threadIdx.x;
  if (i < (long)E * HDS){
    const int e = (int)(i >> 2), h = (int)(i & 3);
    const int s = src ? src[e] : e;
    A[i] = lreluf(SSp[(long)s * HDS + h] + SDp[(long)dst[e] * HDS + h]);
  }
}
__global__ void k_seg_max(const float* __restrict__ A, const int* __restrict__ dst,
                          unsigned* __restrict__ MXp, int E, int H){
  long i = (long)blockIdx.x * blockDim.x + threadIdx.x;
  if (i < (long)E * H){
    const int e = (int)(i / H), h = (int)(i % H);
    atomicMax(&MXp[(long)dst[e] * H + h], ou_enc(A[i]));
  }
}
__global__ void k_seg_expsum(const float* __restrict__ A, const int* __restrict__ dst,
                             const unsigned* __restrict__ MXp, float* __restrict__ DEN,
                             float* __restrict__ Ee, int E, int H){
  long i = (long)blockIdx.x * blockDim.x + threadIdx.x;
  if (i < (long)E * H){
    const int e = (int)(i / H), h = (int)(i % H);
    const long s = (long)dst[e] * H + h;
    const float v = expf(A[i] - ou_dec(MXp[s]));
    Ee[i] = v;
    atomicAdd(&DEN[s], v);
  }
}
__global__ void k_gate_scatter(const float* __restrict__ Ee, const float* __restrict__ DEN,
                               const int* __restrict__ dst, const float* __restrict__ M2,
                               float* __restrict__ Hc, int E){
  long i = (long)blockIdx.x * blockDim.x + threadIdx.x;
  if (i < (long)E * CH){
    const int e = (int)(i >> 7), c = (int)(i & 127);
    const int d = dst[e];
    const float w = Ee[e] / DEN[d];
    atomicAdd(&Hc[(long)d * CH + c], w * M2[i]);
  }
}
__global__ void k_gat_scatter(const float* __restrict__ Ee, const float* __restrict__ DEN,
                              const int* __restrict__ dst, const int* __restrict__ src,
                              const float* __restrict__ Q, float* __restrict__ Oc, int E){
  long i = (long)blockIdx.x * blockDim.x + threadIdx.x;
  if (i < (long)E * HC){
    const int e = (int)(i >> 9), j = (int)(i & 511), h = j >> 7;
    const int d = dst[e];
    const int s = src ? src[e] : e;
    const float w = Ee[(long)e * HDS + h] / DEN[(long)d * HDS + h];
    atomicAdd(&Oc[(long)d * HC + j], w * Q[(long)s * HC + j]);
  }
}
__global__ void k_bias_elu(const float* __restrict__ Hc, const float* __restrict__ bias,
                           float* __restrict__ T, u16* __restrict__ Tb, int N){
  long i = (long)blockIdx.x * blockDim.x + threadIdx.x;
  if (i < (long)N * CH){
    const float v = eluf(Hc[i] + bias[i & 127]);
    T[i] = v; Tb[i] = f2bf(v);
  }
}
__global__ void k_headmean_bias_elu(const float* __restrict__ Oc, const float* __restrict__ bias,
                                    float* __restrict__ T, u16* __restrict__ Tb, int N){
  long i = (long)blockIdx.x * blockDim.x + threadIdx.x;
  if (i < (long)N * CH){
    const int n = (int)(i >> 7), c = (int)(i & 127);
    const float* o = Oc + (long)n * HC + c;
    const float v = eluf(0.25f * (o[0] + o[CH] + o[2 * CH] + o[3 * CH]) + bias[c]);
    T[i] = v; Tb[i] = f2bf(v);
  }
}
__global__ void k_gru(const float* __restrict__ GI, const float* __restrict__ GH,
                      const float* __restrict__ HID, float* __restrict__ OUT, int N){
  long i = (long)blockIdx.x * blockDim.x + threadIdx.x;
  if (i < (long)N * CH){
    const int n = (int)(i >> 7), c = (int)(i & 127);
    const float* gi = GI + (long)n * G3;
    const float* gh = GH + (long)n * G3;
    const float r  = sigmf(gi[c] + gh[c]);
    const float z  = sigmf(gi[CH + c] + gh[CH + c]);
    const float nn = tanhf(gi[2 * CH + c] + r * gh[2 * CH + c]);
    OUT[i] = (1.f - z) * nn + z * HID[i];
  }
}
__global__ void k_add_relu(const float* a, const float* b, float* o, u16* ob, long n){
  long i = (long)blockIdx.x * blockDim.x + threadIdx.x;
  if (i < n){
    const float v = fmaxf(a[i] + b[i], 0.f);
    o[i] = v; if (ob) ob[i] = f2bf(v);
  }
}
__global__ void k_relu_ip_m(float* a, u16* ab, long n){
  long i = (long)blockIdx.x * blockDim.x + threadIdx.x;
  if (i < n){
    const float v = fmaxf(a[i], 0.f);
    a[i] = v; ab[i] = f2bf(v);
  }
}
__global__ void k_seg_feat_sum(const float* __restrict__ X, const int* __restrict__ batch,
                               float* __restrict__ M, int N){
  long i = (long)blockIdx.x * blockDim.x + threadIdx.x;
  if (i < (long)N * CH)
    atomicAdd(&M[(long)batch[i >> 7] * CH + (i & 127)], X[i]);
}
__global__ void k_concat(const float* A, const float* B, float* O, u16* Ob, int N){
  long i = (long)blockIdx.x * blockDim.x + threadIdx.x;
  if (i < (long)N * 2 * CH){
    const int n = (int)(i >> 8), j = (int)(i & 255);
    const float v = (j < CH) ? A[(long)n * CH + j] : B[(long)n * CH + (j - CH)];
    O[i] = v; Ob[i] = f2bf(v);
  }
}
__global__ void k_rowdot(const float* __restrict__ X, const float* __restrict__ W,
                         const float* __restrict__ b, float* __restrict__ Y, int N, int K){
  int n = blockIdx.x * blockDim.x + threadIdx.x;
  if (n < N){
    float s = b[0];
    const float* x = X + (long)n * K;
    for (int k = 0; k < K; ++k) s += x[k] * W[k];
    Y[n] = s;
  }
}

// ---------------- param plumbing ----------------
// params pytree flattened alphabetically, leaves after the 14 array inputs:
//  14 lin_la.W 15 lin_la.b 16 lin_pa.W 17 lin_pa.b
//  gate (13): W1,W2,att_l,att_r,bias,gru.Wh,gru.Wi,gru.bh,gru.bi,lin_dst.W,lin_dst.b,lin_src.W,lin_src.b
//  gat  (9):  Wd,Ws,att_d,att_s,bias,gru.Wh,gru.Wi,gru.bh,gru.bi
//  18 ll0 | 31 ll1 | 40 ll2 | 49 lp0 | 62 lp1 | 71 lp2 | 80..85 mlp
//  86 mol_la | 95 mol_pa | 104 pl0 | 117 pl1 | 126 pl2 | 135 pp0 | 148 pp1 | 157 pp2
struct GateP { const float *W1,*W2,*att_l,*att_r,*bias,*gWh,*gWi,*gbh,*gbi,*ldW,*ldb,*lsW,*lsb; };
struct GatP  { const float *Wd,*Ws,*att_d,*att_s,*bias,*gWh,*gWi,*gbh,*gbi; };
struct GateW { u16 *lsT,*ldT,*W1T,*W2T,*WiT,*WhT; };
struct GatW  { u16 *WsT,*WdT,*WiT,*WhT; };

static GateP loadGate(void* const* d, int i){
  GateP p; const float** f = (const float**)&p;
  for (int j = 0; j < 13; ++j) f[j] = (const float*)d[i + j];
  return p;
}
static GatP loadGat(void* const* d, int i){
  GatP p; const float** f = (const float**)&p;
  for (int j = 0; j < 9; ++j) f[j] = (const float*)d[i + j];
  return p;
}
static inline int gceil(long n, int b){ return (int)((n + b - 1) / b); }

extern "C" void kernel_launch(void* const* d_in, const int* in_sizes, int n_in,
                              void* d_out, int out_size, void* d_ws, size_t ws_size,
                              hipStream_t stream)
{
  (void)in_sizes; (void)n_in; (void)out_size; (void)ws_size;
  const float* x_pa = (const float*)d_in[0];
  const float* x_la = (const float*)d_in[1];
  const int* ei_pp  = (const int*)d_in[2];
  const int* ei_ll  = (const int*)d_in[3];
  const int* lp_src = (const int*)d_in[4];
  const int* lp_dst = (const int*)d_in[5];
  const int* pl_src = (const int*)d_in[6];
  const int* pl_dst = (const int*)d_in[7];
  const float* ea_pp = (const float*)d_in[8];
  const float* ea_ll = (const float*)d_in[9];
  const float* ea_lp = (const float*)d_in[10];
  const float* ea_pl = (const float*)d_in[11];
  const int* batch_pa = (const int*)d_in[12];
  const int* batch_la = (const int*)d_in[13];

  const float* lin_laW = (const float*)d_in[14]; const float* lin_lab = (const float*)d_in[15];
  const float* lin_paW = (const float*)d_in[16]; const float* lin_pab = (const float*)d_in[17];
  const GateP ll0 = loadGate(d_in, 18);
  const GatP  ll1 = loadGat(d_in, 31), ll2 = loadGat(d_in, 40);
  const GateP lp0 = loadGate(d_in, 49);
  const GatP  lp1 = loadGat(d_in, 62), lp2 = loadGat(d_in, 71);
  const float* mlp0W = (const float*)d_in[80]; const float* mlp0b = (const float*)d_in[81];
  const float* mlp1W = (const float*)d_in[82]; const float* mlp1b = (const float*)d_in[83];
  const float* mlp2W = (const float*)d_in[84]; const float* mlp2b = (const float*)d_in[85];
  const GatP  mol_la = loadGat(d_in, 86), mol_pa = loadGat(d_in, 95);
  const GateP pl0 = loadGate(d_in, 104);
  const GatP  pl1 = loadGat(d_in, 117), pl2 = loadGat(d_in, 126);
  const GateP pp0 = loadGate(d_in, 135);
  const GatP  pp1 = loadGat(d_in, 148), pp2 = loadGat(d_in, 157);

  // ---------------- workspace bump allocator ----------------
  char* wp = (char*)d_ws; size_t off = 0;
  auto allocB = [&](size_t bytes)->void*{
    off = (off + 255) & ~(size_t)255;
    void* p = (void*)(wp + off);
    off += bytes;
    return p;
  };
  auto allocF = [&](size_t n)->float*{ return (float*)allocB(n * 4); };
  auto allocH = [&](size_t n)->u16*  { return (u16*)allocB(n * 2); };

  // f32 activations
  float* P0  = allocF((size_t)NPA * CH);
  float* P1  = allocF((size_t)NPA * CH);
  float* GB  = allocF((size_t)NPA * CH);
  float* L0  = allocF((size_t)NLA * CH);
  float* L1  = allocF((size_t)NLA * CH);
  float* LB  = allocF((size_t)NLA * CH);
  float* X1A = allocF((size_t)NPA * CH);
  float* X1B = allocF((size_t)NPA * CH);
  float* LX1 = allocF((size_t)NLA * CH);
  float* TB  = allocF((size_t)NPA * CH);
  float* SS  = allocF((size_t)NPA * HDS);
  float* SD  = allocF((size_t)NPA * HDS);
  float* ALOG = allocF((size_t)E_PP * HDS);
  float* ELOG = allocF((size_t)E_PP * HDS);
  unsigned* MX = (unsigned*)allocF((size_t)NPA * HDS);
  float* DEN = allocF((size_t)NPA * HDS);
  float* BIG0 = allocF((size_t)NPA * HC);   // m2 / QS / GI
  float* BIG1 = allocF((size_t)NPA * HC);   // QD / GH
  float* BIG2 = allocF((size_t)NPA * HC);   // Hacc / Oacc
  float* QDm  = allocF((size_t)NBC * HC);
  float* OACCm= allocF((size_t)NBC * HC);
  float* GIm  = allocF((size_t)NBC * G3);
  float* GHm  = allocF((size_t)NBC * G3);
  float* MOL0 = allocF((size_t)NBC * CH);
  float* MOL1 = allocF((size_t)NBC * CH);
  float* MOL2 = allocF((size_t)NBC * CH);
  float* MOL3 = allocF((size_t)NBC * CH);
  float* YPA  = allocF((size_t)NBC * CH);
  float* YLA  = allocF((size_t)NBC * CH);
  float* XC   = allocF((size_t)NBC * 2 * CH);
  float* H1   = allocF((size_t)NBC * CH);
  float* H2   = allocF((size_t)NBC * 64);
  // bf16 mirrors (GEMM A-operands)
  u16* XPAb = allocH((size_t)NPA * KINP);
  u16* XLAb = allocH((size_t)NLA * KINP);
  u16* X1Ab = allocH((size_t)NPA * CH);
  u16* X1Bb = allocH((size_t)NPA * CH);
  u16* LX1b = allocH((size_t)NLA * CH);
  u16* TBb  = allocH((size_t)NPA * CH);
  u16* P0b  = allocH((size_t)NPA * CH);
  u16* P1b  = allocH((size_t)NPA * CH);
  u16* L0b  = allocH((size_t)NLA * CH);
  u16* L1b  = allocH((size_t)NLA * CH);
  u16* MOL0b = allocH((size_t)NBC * CH);
  u16* MOL1b = allocH((size_t)NBC * CH);
  u16* MOL2b = allocH((size_t)NBC * CH);
  u16* MOL3b = allocH((size_t)NBC * CH);
  u16* XCb  = allocH((size_t)NBC * 2 * CH);
  u16* H1b  = allocH((size_t)NBC * CH);

  // ---------------- host helpers ----------------
  auto gemm = [&](const u16* Xb, const u16* WT, const float* bias, float* Y, u16* Yb,
                  int N, int Kp, int M, int act){
    dim3 g(N / 16, M / 64);
    k_gemm_bf<<<g, 32, 0, stream>>>(Xb, WT, bias, Y, Yb, Kp, M, act);
  };
  auto fillf = [&](float* p, float v, long n){
    k_fill_f32<<<gceil(n, 256), 256, 0, stream>>>(p, v, n);
  };
  auto fillu = [&](unsigned* p, unsigned v, long n){
    k_fill_u32<<<gceil(n, 256), 256, 0, stream>>>(p, v, n);
  };
  auto packW = [&](const float* W, int K, int M, int Kp)->u16*{
    u16* p = allocH((size_t)M * Kp);
    k_pack_wT<<<gceil((long)M * Kp, 256), 256, 0, stream>>>(W, p, K, M, Kp);
    return p;
  };
  auto packGate = [&](const GateP& g)->GateW{
    GateW w;
    w.lsT = packW(g.lsW, KIN, CH, KINP);
    w.ldT = packW(g.ldW, KIN, CH, KINP);
    w.W1T = packW(g.W1, KE, CH, KEP);
    w.W2T = packW(g.W2, CH, CH, CH);
    w.WiT = packW(g.gWi, CH, G3, CH);
    w.WhT = packW(g.gWh, CH, G3, CH);
    return w;
  };
  auto packGat = [&](const GatP& g)->GatW{
    GatW w;
    w.WsT = packW(g.Ws, CH, HC, CH);
    w.WdT = packW(g.Wd, CH, HC, CH);
    w.WiT = packW(g.gWi, CH, G3, CH);
    w.WhT = packW(g.gWh, CH, G3, CH);
    return w;
  };

  // pack all weights + padded bf16 inputs
  const GateW wpp0 = packGate(pp0), wll0 = packGate(ll0), wlp0 = packGate(lp0), wpl0 = packGate(pl0);
  const GatW wpp[2] = {packGat(pp1), packGat(pp2)};
  const GatW wll[2] = {packGat(ll1), packGat(ll2)};
  const GatW wlp[2] = {packGat(lp1), packGat(lp2)};
  const GatW wpl[2] = {packGat(pl1), packGat(pl2)};
  const GatW wmolpa = packGat(mol_pa), wmolla = packGat(mol_la);
  u16* linpaT = packW(lin_paW, CH, CH, CH);
  u16* linlaT = packW(lin_laW, CH, CH, CH);
  u16* mlp0T  = packW(mlp0W, 2 * CH, CH, 2 * CH);
  u16* mlp1T  = packW(mlp1W, CH, 64, CH);
  k_pack_rows<<<gceil((long)NPA * KINP, 256), 256, 0, stream>>>(x_pa, XPAb, KIN, KINP, NPA);
  k_pack_rows<<<gceil((long)NLA * KINP, 256), 256, 0, stream>>>(x_la, XLAb, KIN, KINP, NLA);

  // GATEConv + GRU (layer 0). xdb==nullptr -> x1d aliases x1s.
  auto gate_conv = [&](const GateP& g, const GateW& w,
                       const u16* xsb, int Ns, const u16* xdb, int Nd,
                       float* x1s, u16* x1sb, float* x1d, u16* x1db,
                       const int* src, const int* dst, const float* ea, int E,
                       float* out){
    (void)Ns;
    gemm(xsb, w.lsT, g.lsb, x1s, x1sb, Ns, KINP, CH, 1);
    if (xdb) gemm(xdb, w.ldT, g.ldb, x1d, x1db, Nd, KINP, CH, 1);
    else { x1d = x1s; x1db = x1sb; }
    k_node_dot<<<gceil(Nd, 256), 256, 0, stream>>>(x1d, g.att_r, SS, Nd);
    k_gate_edge<<<E / 16, 256, 0, stream>>>(x1sb, src, dst, ea, w.W1T, w.W2T,
                                            g.att_l, SS, ALOG, BIG0);
    fillu(MX, 0u, Nd);
    k_seg_max<<<gceil((long)E, 256), 256, 0, stream>>>(ALOG, dst, MX, E, 1);
    fillf(DEN, 0.f, Nd);
    k_seg_expsum<<<gceil((long)E, 256), 256, 0, stream>>>(ALOG, dst, MX, DEN, ELOG, E, 1);
    fillf(BIG2, 0.f, (long)Nd * CH);
    k_gate_scatter<<<gceil((long)E * CH, 256), 256, 0, stream>>>(ELOG, DEN, dst, BIG0, BIG2, E);
    k_bias_elu<<<gceil((long)Nd * CH, 256), 256, 0, stream>>>(BIG2, g.bias, TB, TBb, Nd);
    gemm(TBb,  w.WiT, g.gbi, BIG0, nullptr, Nd, CH, G3, 0);
    gemm(x1db, w.WhT, g.gbh, BIG1, nullptr, Nd, CH, G3, 0);
    k_gru<<<gceil((long)Nd * CH, 256), 256, 0, stream>>>(BIG0, BIG1, x1d, out, Nd);
  };

  // GAT + GRU (layers 1..)
  auto gat_conv = [&](const GatP& g, const GatW& w,
                      const u16* xsb, int Ns, const float* xd, const u16* xdb, int Nd,
                      const int* src, const int* dst, int E,
                      float* out){
    gemm(xsb, w.WsT, nullptr, BIG0, nullptr, Ns, CH, HC, 0);
    gemm(xdb, w.WdT, nullptr, BIG1, nullptr, Nd, CH, HC, 0);
    k_att_dot<<<gceil((long)Ns * HDS, 256), 256, 0, stream>>>(BIG0, g.att_s, SS, Ns);
    k_att_dot<<<gceil((long)Nd * HDS, 256), 256, 0, stream>>>(BIG1, g.att_d, SD, Nd);
    k_gat_logit<<<gceil((long)E * HDS, 256), 256, 0, stream>>>(SS, SD, src, dst, ALOG, E);
    fillu(MX, 0u, (long)Nd * HDS);
    k_seg_max<<<gceil((long)E * HDS, 256), 256, 0, stream>>>(ALOG, dst, MX, E, HDS);
    fillf(DEN, 0.f, (long)Nd * HDS);
    k_seg_expsum<<<gceil((long)E * HDS, 256), 256, 0, stream>>>(ALOG, dst, MX, DEN, ELOG, E, HDS);
    fillf(BIG2, 0.f, (long)Nd * HC);
    k_gat_scatter<<<gceil((long)E * HC, 256), 256, 0, stream>>>(ELOG, DEN, dst, src, BIG0, BIG2, E);
    k_headmean_bias_elu<<<gceil((long)Nd * CH, 256), 256, 0, stream>>>(BIG2, g.bias, TB, TBb, Nd);
    gemm(TBb, w.WiT, g.gbi, BIG0, nullptr, Nd, CH, G3, 0);
    gemm(xdb, w.WhT, g.gbh, BIG1, nullptr, Nd, CH, G3, 0);
    k_gru<<<gceil((long)Nd * CH, 256), 256, 0, stream>>>(BIG0, BIG1, xd, out, Nd);
  };

  // ---------------- layer 0 (GATEConv hetero) ----------------
  gate_conv(pp0, wpp0, XPAb, NPA, nullptr, NPA, X1A, X1Ab, X1A, X1Ab,
            ei_pp, ei_pp + E_PP, ea_pp, E_PP, P1);
  gate_conv(lp0, wlp0, XLAb, NLA, XPAb, NPA, LX1, LX1b, X1B, X1Bb,
            lp_src, lp_dst, ea_lp, E_LP, GB);
  k_add_relu<<<gceil((long)NPA * CH, 256), 256, 0, stream>>>(P1, GB, P0, P0b, (long)NPA * CH);
  gate_conv(ll0, wll0, XLAb, NLA, nullptr, NLA, LX1, LX1b, LX1, LX1b,
            ei_ll, ei_ll + E_LL, ea_ll, E_LL, L1);
  gate_conv(pl0, wpl0, XPAb, NPA, XLAb, NLA, X1A, X1Ab, LX1, LX1b,
            pl_src, pl_dst, ea_pl, E_PL, LB);
  k_add_relu<<<gceil((long)NLA * CH, 256), 256, 0, stream>>>(L1, LB, L0, L0b, (long)NLA * CH);

  // ---------------- layers 1..2 (GAT hetero) ----------------
  const GatP gpp[2] = {pp1, pp2}, gll[2] = {ll1, ll2}, glp[2] = {lp1, lp2}, gpl[2] = {pl1, pl2};
  float *paC = P0, *laC = L0, *paN = P1, *laN = L1;
  u16 *paCb = P0b, *laCb = L0b, *paNb = P1b, *laNb = L1b;
  for (int l = 0; l < 2; ++l){
    gat_conv(gpp[l], wpp[l], paCb, NPA, paC, paCb, NPA, ei_pp, ei_pp + E_PP, E_PP, paN);
    gat_conv(glp[l], wlp[l], laCb, NLA, paC, paCb, NPA, lp_src, lp_dst, E_LP, GB);
    k_add_relu<<<gceil((long)NPA * CH, 256), 256, 0, stream>>>(paN, GB, paN, paNb, (long)NPA * CH);
    gat_conv(gll[l], wll[l], laCb, NLA, laC, laCb, NLA, ei_ll, ei_ll + E_LL, E_LL, laN);
    gat_conv(gpl[l], wpl[l], paCb, NPA, laC, laCb, NLA, pl_src, pl_dst, E_PL, LB);
    k_add_relu<<<gceil((long)NLA * CH, 256), 256, 0, stream>>>(laN, LB, laN, laNb, (long)NLA * CH);
    float* tf; u16* th;
    tf = paC; paC = paN; paN = tf;  th = paCb; paCb = paNb; paNb = th;
    tf = laC; laC = laN; laN = tf;  th = laCb; laCb = laNb; laNb = th;
  }

  // ---------------- molecular pooling ----------------
  fillf(MOL0, 0.f, (long)NBC * CH);
  k_seg_feat_sum<<<gceil((long)NPA * CH, 256), 256, 0, stream>>>(paC, batch_pa, MOL0, NPA);
  k_relu_ip_m<<<gceil((long)NBC * CH, 256), 256, 0, stream>>>(MOL0, MOL0b, (long)NBC * CH);
  fillf(MOL2, 0.f, (long)NBC * CH);
  k_seg_feat_sum<<<gceil((long)NLA * CH, 256), 256, 0, stream>>>(laC, batch_la, MOL2, NLA);
  k_relu_ip_m<<<gceil((long)NBC * CH, 256), 256, 0, stream>>>(MOL2, MOL2b, (long)NBC * CH);

  auto mol_phase = [&](const GatP& g, const GatW& w, const u16* atomsb, int Na,
                       const int* batch, float* m0, u16* m0b, float* m1, u16* m1b,
                       float** outF, u16** outB){
    gemm(atomsb, w.WsT, nullptr, BIG0, nullptr, Na, CH, HC, 0);    // QS fixed over timesteps
    k_att_dot<<<gceil((long)Na * HDS, 256), 256, 0, stream>>>(BIG0, g.att_s, SS, Na);
    float *mc = m0, *mn = m1; u16 *mcb = m0b, *mnb = m1b;
    for (int t = 0; t < 3; ++t){
      gemm(mcb, w.WdT, nullptr, QDm, nullptr, NBC, CH, HC, 0);
      k_att_dot<<<gceil((long)NBC * HDS, 256), 256, 0, stream>>>(QDm, g.att_d, SD, NBC);
      k_gat_logit<<<gceil((long)Na * HDS, 256), 256, 0, stream>>>(SS, SD, nullptr, batch, ALOG, Na);
      fillu(MX, 0u, (long)NBC * HDS);
      k_seg_max<<<gceil((long)Na * HDS, 256), 256, 0, stream>>>(ALOG, batch, MX, Na, HDS);
      fillf(DEN, 0.f, (long)NBC * HDS);
      k_seg_expsum<<<gceil((long)Na * HDS, 256), 256, 0, stream>>>(ALOG, batch, MX, DEN, ELOG, Na, HDS);
      fillf(OACCm, 0.f, (long)NBC * HC);
      k_gat_scatter<<<gceil((long)Na * HC, 256), 256, 0, stream>>>(ELOG, DEN, batch, nullptr, BIG0, OACCm, Na);
      k_headmean_bias_elu<<<gceil((long)NBC * CH, 256), 256, 0, stream>>>(OACCm, g.bias, TB, TBb, NBC);
      gemm(TBb, w.WiT, g.gbi, GIm, nullptr, NBC, CH, G3, 0);
      gemm(mcb, w.WhT, g.gbh, GHm, nullptr, NBC, CH, G3, 0);
      k_gru<<<gceil((long)NBC * CH, 256), 256, 0, stream>>>(GIm, GHm, mc, mn, NBC);
      k_relu_ip_m<<<gceil((long)NBC * CH, 256), 256, 0, stream>>>(mn, mnb, (long)NBC * CH);
      float* tf = mc; mc = mn; mn = tf;
      u16* th = mcb; mcb = mnb; mnb = th;
    }
    *outF = mc; *outB = mcb;
  };
  float *molPA, *molLA; u16 *molPAb, *molLAb;
  mol_phase(mol_pa, wmolpa, paCb, NPA, batch_pa, MOL0, MOL0b, MOL1, MOL1b, &molPA, &molPAb);
  mol_phase(mol_la, wmolla, laCb, NLA, batch_la, MOL2, MOL2b, MOL3, MOL3b, &molLA, &molLAb);

  // ---------------- readout MLP ----------------
  gemm(molPAb, linpaT, lin_pab, YPA, nullptr, NBC, CH, CH, 0);
  gemm(molLAb, linlaT, lin_lab, YLA, nullptr, NBC, CH, CH, 0);
  k_concat<<<gceil((long)NBC * 2 * CH, 256), 256, 0, stream>>>(YPA, YLA, XC, XCb, NBC);
  gemm(XCb, mlp0T, mlp0b, H1, H1b, NBC, 2 * CH, CH, 2);
  gemm(H1b, mlp1T, mlp1b, H2, nullptr, NBC, CH, 64, 2);
  k_rowdot<<<gceil((long)NBC, 256), 256, 0, stream>>>(H2, mlp2W, mlp2b, (float*)d_out, NBC, 64);
  (void)molPA; (void)molLA;
}
